// SRCMBlock_8211977470536
// MI455X (gfx1250) — compile-verified, matrix operand order checked by reading, not run on hardware
//
#include <hip/hip_runtime.h>
#include <cstddef>
#include <cstdint>

// ---------------- problem constants ----------------
#define CCH    64              // channels C
#define DSTATE 16
#define DCONV  4
#define DIN    128             // d_inner
#define DTRANK 4
#define NGRP   8
#define HWSZ   (96 * 96)       // 9216 = L
#define BATCH  2
#define LTOK   HWSZ
#define MTOK   (BATCH * LTOK)  // 18432 tokens
#define XPAD   64              // x_proj cols padded 36 -> 64 (pairable 16-tiles)
#define LN_EPS 1e-5f

typedef __bf16 bf16_t;
typedef __attribute__((ext_vector_type(16))) __bf16 v16bf;
typedef __attribute__((ext_vector_type(8)))  __bf16 v8bf;
typedef __attribute__((ext_vector_type(8)))  float  v8f;

// ---- fragment loaders -------------------------------------------------
// A fragment (fp32 -> bf16): lane holds K {lo..lo+7, lo+16..lo+23}; both
// chunks are 8 contiguous floats = 2x float4 (32B) -> v_cvt_pk_bf16_f32.
__device__ __forceinline__ v16bf pack16(const float* __restrict__ p) {
  float4 x0 = *(const float4*)(p);
  float4 x1 = *(const float4*)(p + 4);
  float4 x2 = *(const float4*)(p + 16);
  float4 x3 = *(const float4*)(p + 20);
  v16bf r;
  r[0] = (__bf16)x0.x;  r[1] = (__bf16)x0.y;  r[2] = (__bf16)x0.z;  r[3] = (__bf16)x0.w;
  r[4] = (__bf16)x1.x;  r[5] = (__bf16)x1.y;  r[6] = (__bf16)x1.z;  r[7] = (__bf16)x1.w;
  r[8] = (__bf16)x2.x;  r[9] = (__bf16)x2.y;  r[10] = (__bf16)x2.z; r[11] = (__bf16)x2.w;
  r[12] = (__bf16)x3.x; r[13] = (__bf16)x3.y; r[14] = (__bf16)x3.z; r[15] = (__bf16)x3.w;
  return r;
}
__device__ __forceinline__ v16bf pack16sum(const float* __restrict__ p,
                                           const float* __restrict__ q) {
  float4 x0 = *(const float4*)(p);      float4 y0 = *(const float4*)(q);
  float4 x1 = *(const float4*)(p + 4);  float4 y1 = *(const float4*)(q + 4);
  float4 x2 = *(const float4*)(p + 16); float4 y2 = *(const float4*)(q + 16);
  float4 x3 = *(const float4*)(p + 20); float4 y3 = *(const float4*)(q + 20);
  v16bf r;
  r[0] = (__bf16)(x0.x + y0.x);  r[1] = (__bf16)(x0.y + y0.y);
  r[2] = (__bf16)(x0.z + y0.z);  r[3] = (__bf16)(x0.w + y0.w);
  r[4] = (__bf16)(x1.x + y1.x);  r[5] = (__bf16)(x1.y + y1.y);
  r[6] = (__bf16)(x1.z + y1.z);  r[7] = (__bf16)(x1.w + y1.w);
  r[8] = (__bf16)(x2.x + y2.x);  r[9] = (__bf16)(x2.y + y2.y);
  r[10] = (__bf16)(x2.z + y2.z); r[11] = (__bf16)(x2.w + y2.w);
  r[12] = (__bf16)(x3.x + y3.x); r[13] = (__bf16)(x3.y + y3.y);
  r[14] = (__bf16)(x3.z + y3.z); r[15] = (__bf16)(x3.w + y3.w);
  return r;
}
// B fragment (bf16 weights): two contiguous 8-bf16 (16B) chunks -> b128 loads.
__device__ __forceinline__ v16bf loadw(const bf16_t* __restrict__ p) {
  v8bf lo = *(const v8bf*)(p);
  v8bf hi = *(const v8bf*)(p + 16);
  return __builtin_shufflevector(lo, hi, 0, 1, 2, 3, 4, 5, 6, 7,
                                 8, 9, 10, 11, 12, 13, 14, 15);
}

// =====================================================================
// bf16 WMMA GEMM:  C[M,N] = (A (+A2))[M,K] * W[N,K]^T  (+bias)
// One 16(M)x32(N) block per wave32: one shared A fragment feeding two
// independent v_wmma_f32_16x16x32_bf16 accumulator chains; K fully
// unrolled (templated).  Optional channel-major store + residual.
// =====================================================================
template <int K, bool HAS_A2>
__global__ void k_gemm_bf16(const float* __restrict__ A,
                            const float* __restrict__ A2,
                            const bf16_t* __restrict__ W,
                            float* __restrict__ Cout,
                            const float* __restrict__ bias,
                            const float* __restrict__ resid,
                            int N, int cmStore) {
  const int lane  = threadIdx.x & 31;
  const int wave  = blockIdx.x * (blockDim.x >> 5) + (threadIdx.x >> 5);
  const int nPair = N >> 5;
  const int m0    = (wave / nPair) << 4;
  const int n0    = (wave % nPair) << 5;
  const int r     = lane & 15;
  // ISA 7.12.2: lanes 0-15 hold K {0..7,16..23}, lanes 16-31 {8..15,24..31}
  const int lo    = (lane & 16) ? 8 : 0;

  const float*  ap  = A + (size_t)(m0 + r) * K + lo;
  const float*  ap2 = HAS_A2 ? (A2 + (size_t)(m0 + r) * K + lo) : nullptr;
  const bf16_t* wp0 = W + (size_t)(n0 + r) * K + lo;
  const bf16_t* wp1 = wp0 + (size_t)16 * K;

  v8f acc0 = {0.f, 0.f, 0.f, 0.f, 0.f, 0.f, 0.f, 0.f};
  v8f acc1 = {0.f, 0.f, 0.f, 0.f, 0.f, 0.f, 0.f, 0.f};

#pragma unroll
  for (int k0 = 0; k0 < K; k0 += 32) {
    v16bf a  = HAS_A2 ? pack16sum(ap + k0, ap2 + k0) : pack16(ap + k0);
    v16bf b0 = loadw(wp0 + k0);
    v16bf b1 = loadw(wp1 + k0);
    acc0 = __builtin_amdgcn_wmma_f32_16x16x32_bf16(false, a, false, b0,
                                                   (short)0, acc0, false, false);
    acc1 = __builtin_amdgcn_wmma_f32_16x16x32_bf16(false, a, false, b1,
                                                   (short)0, acc1, false, false);
  }

  // C/D layout: VGPR v -> M = m0 + v + 8*(lane>=16), N = n0 + (lane&15)
  const int rbase = m0 + ((lane & 16) ? 8 : 0);
#pragma unroll
  for (int t = 0; t < 2; ++t) {
    const v8f&  acc = t ? acc1 : acc0;
    const int   col = n0 + t * 16 + r;
    const float bvv = bias ? bias[col] : 0.f;
#pragma unroll
    for (int v = 0; v < 8; ++v) {
      int   row = rbase + v;
      float val = acc[v] + bvv;
      if (cmStore) {                     // token-major -> (B,C,HW)
        int    bb = (row >= LTOK) ? 1 : 0;
        int    ll = row - bb * LTOK;
        size_t o  = ((size_t)bb * CCH + col) * HWSZ + ll;
        if (resid) val += resid[o];
        Cout[o] = val;
      } else {
        Cout[(size_t)row * N + col] = val;
      }
    }
  }
}

// =====================================================================
// Token LayerNorm over C=64.  One wave32 per token, 2 channels/lane,
// wave32 shuffle reduction.  val = (Yrow? Yrow : 0) + s * Xcm.
// =====================================================================
__global__ void k_ln_tok(const float* __restrict__ Xcm,
                         const float* __restrict__ Yrow,
                         const float* __restrict__ skipPtr,
                         const float* __restrict__ g,
                         const float* __restrict__ bta,
                         float* __restrict__ out) {
  const int lane = threadIdx.x & 31;
  const int tk   = blockIdx.x * (blockDim.x >> 5) + (threadIdx.x >> 5);
  const int b    = (tk >= LTOK) ? 1 : 0;
  const int l    = tk - b * LTOK;
  const float s  = skipPtr ? skipPtr[0] : 1.0f;

  float v0 = s * Xcm[((size_t)b * CCH + lane) * HWSZ + l];
  float v1 = s * Xcm[((size_t)b * CCH + lane + 32) * HWSZ + l];
  if (Yrow) {
    v0 += Yrow[(size_t)tk * CCH + lane];
    v1 += Yrow[(size_t)tk * CCH + lane + 32];
  }
  float sm = v0 + v1;
  float sq = v0 * v0 + v1 * v1;
#pragma unroll
  for (int m = 16; m >= 1; m >>= 1) {
    sm += __shfl_xor(sm, m, 32);
    sq += __shfl_xor(sq, m, 32);
  }
  const float mean = sm * (1.0f / 64.0f);
  const float var  = sq * (1.0f / 64.0f) - mean * mean;
  const float inv  = rsqrtf(var + LN_EPS);
  out[(size_t)tk * CCH + lane]      = (v0 - mean) * inv * g[lane] + bta[lane];
  out[(size_t)tk * CCH + lane + 32] = (v1 - mean) * inv * g[lane + 32] + bta[lane + 32];
}

// =====================================================================
// GroupNorm(8 groups) + ReLU on (B,C,HW).  One block per (b,group).
// =====================================================================
__global__ void k_gn_relu(const float* __restrict__ X,
                          const float* __restrict__ g,
                          const float* __restrict__ bta,
                          float* __restrict__ out) {
  __shared__ float s1[256], s2[256];
  const int    bg   = blockIdx.x;
  const int    b    = bg / NGRP, gr = bg % NGRP;
  const int    CG   = CCH / NGRP;          // 8
  const size_t base = ((size_t)b * CCH + gr * CG) * HWSZ;
  const int    NEL  = CG * HWSZ;           // 73728
  float a = 0.f, q = 0.f;
  for (int i = threadIdx.x; i < NEL; i += 256) {
    float v = X[base + i];
    a += v; q += v * v;
  }
  s1[threadIdx.x] = a; s2[threadIdx.x] = q;
  __syncthreads();
  for (int st = 128; st >= 1; st >>= 1) {
    if ((int)threadIdx.x < st) {
      s1[threadIdx.x] += s1[threadIdx.x + st];
      s2[threadIdx.x] += s2[threadIdx.x + st];
    }
    __syncthreads();
  }
  const float mean = s1[0] / (float)NEL;
  const float var  = s2[0] / (float)NEL - mean * mean;
  const float inv  = rsqrtf(var + LN_EPS);
  for (int i = threadIdx.x; i < NEL; i += 256) {
    int   c = gr * CG + i / HWSZ;
    float v = (X[base + i] - mean) * inv * g[c] + bta[c];
    out[base + i] = v > 0.f ? v : 0.f;
  }
}

// =====================================================================
// Causal depthwise conv (K=4) + SiLU, reading x = xz[...,:128].
// dir=+1: y[l]=b+sum_j w[j]x[l+j-3]   dir=-1: y[l]=b+sum_j w[j]x[l+3-j]
// =====================================================================
__global__ void k_conv_silu(const float* __restrict__ xz,
                            const float* __restrict__ w,
                            const float* __restrict__ bias,
                            float* __restrict__ xc, int dir) {
  const int idx = blockIdx.x * blockDim.x + threadIdx.x;  // MTOK*DIN exact
  const int d   = idx & (DIN - 1);
  const int tk  = idx >> 7;
  const int b   = (tk >= LTOK) ? 1 : 0;
  const int l   = tk - b * LTOK;
  float s = bias[d];
#pragma unroll
  for (int j = 0; j < DCONV; ++j) {
    int off = (dir > 0) ? (l + j - 3) : (l + 3 - j);
    if (off >= 0 && off < LTOK)
      s += w[d * DCONV + j] * xz[((size_t)(b * LTOK + off)) * 256 + d];
  }
  xc[(size_t)tk * DIN + d] = s / (1.0f + __expf(-s));    // SiLU
}

// =====================================================================
// Selective scan, both directions in one launch: grid (BATCH, 2).
// 512 threads: d = tid>>2 (128 channels), 4 lanes per d each owning 4 of
// the N=16 states in registers.  Per-token x_dbl (dt|B|C; cols 0..35 of
// the 64-wide padded row) broadcast through LDS; y reduced with wave32
// shuffles; next token's operands warmed with global_prefetch_b8.
// =====================================================================
struct ScanArgs {
  const float* xc[2];
  const float* xdbl[2];
  const float* dtw[2];
  const float* dtb[2];
  const float* An[2];    // A = -exp(A_log), (128,16)
  const float* Dp[2];
  const float* xz;       // z = xz[...,128:]
  float*       yo[2];
};

__global__ void k_scan(ScanArgs sa) {
  const int b   = blockIdx.x;
  const int dir = blockIdx.y;          // 0 fwd, 1 bwd
  const int d   = threadIdx.x >> 2;
  const int g   = threadIdx.x & 3;

  const float* xc   = sa.xc[dir];
  const float* xdbl = sa.xdbl[dir];
  const float* An   = sa.An[dir] + d * DSTATE + g * 4;
  float*       yo   = sa.yo[dir];

  const float dtw0 = sa.dtw[dir][d * 4 + 0];
  const float dtw1 = sa.dtw[dir][d * 4 + 1];
  const float dtw2 = sa.dtw[dir][d * 4 + 2];
  const float dtw3 = sa.dtw[dir][d * 4 + 3];
  const float dtb  = sa.dtb[dir][d];
  const float Dd   = sa.Dp[dir][d];
  const float a0 = An[0], a1 = An[1], a2 = An[2], a3 = An[3];
  float h0 = 0.f, h1 = 0.f, h2 = 0.f, h3 = 0.f;

  __shared__ float sd[40];

  for (int t = 0; t < LTOK; ++t) {
    const int    l   = dir ? (LTOK - 1 - t) : t;
    const size_t tkl = (size_t)b * LTOK + l;
    if (threadIdx.x < 36) sd[threadIdx.x] = xdbl[tkl * XPAD + threadIdx.x];
    __syncthreads();

    const float u     = xc[tkl * DIN + d];
    float       dtv   = dtb + dtw0 * sd[0] + dtw1 * sd[1] + dtw2 * sd[2] + dtw3 * sd[3];
    const float delta = (dtv > 20.f) ? dtv : log1pf(__expf(dtv));   // softplus
    const float du    = delta * u;
    const int   nb    = g * 4;
    float y = 0.f, e;
    e = __expf(delta * a0); h0 = e * h0 + du * sd[4 + nb + 0]; y += h0 * sd[20 + nb + 0];
    e = __expf(delta * a1); h1 = e * h1 + du * sd[4 + nb + 1]; y += h1 * sd[20 + nb + 1];
    e = __expf(delta * a2); h2 = e * h2 + du * sd[4 + nb + 2]; y += h2 * sd[20 + nb + 2];
    e = __expf(delta * a3); h3 = e * h3 + du * sd[4 + nb + 3]; y += h3 * sd[20 + nb + 3];
    y += __shfl_xor(y, 1, 32);
    y += __shfl_xor(y, 2, 32);
    if (g == 0) {
      const float z = sa.xz[tkl * 256 + DIN + d];
      yo[tkl * DIN + d] = (y + u * Dd) * (z / (1.f + __expf(-z)));
    }
    if (t + 1 < LTOK) {                 // warm next token's lines
      const int ln = dir ? (l - 1) : (l + 1);
      __builtin_prefetch(&xc[((size_t)b * LTOK + ln) * DIN + d], 0, 1);
      if (threadIdx.x < 36)
        __builtin_prefetch(&xdbl[((size_t)b * LTOK + ln) * XPAD + threadIdx.x], 0, 1);
    }
    __syncthreads();
  }
}

// =====================================================================
// Weight prep: fp32 (rows,cols) -> bf16 (dstRows,cols), zero padded rows.
// =====================================================================
__global__ void k_pad_bf16(const float* __restrict__ src, bf16_t* __restrict__ dst,
                           int rows, int cols, int dstRows) {
  const int i   = blockIdx.x * blockDim.x + threadIdx.x;
  const int tot = dstRows * cols;
  if (i >= tot) return;
  const int r = i / cols;
  dst[i] = (bf16_t)((r < rows) ? src[i] : 0.f);
}

__global__ void k_negexp(const float* __restrict__ src, float* __restrict__ dst, int n) {
  const int i = blockIdx.x * blockDim.x + threadIdx.x;
  if (i < n) dst[i] = -__expf(src[i]);
}

// =====================================================================
// Host side
// =====================================================================
struct LayerP {
  const float *ln_g, *ln_b, *skip, *proj_w, *proj_b, *in_w, *out_w;
  const float *conv_w, *conv_b, *xproj_w, *dt_w, *dt_b, *A_log, *D;
  const float *conv_w_b, *conv_b_b, *xproj_w_b, *dt_w_b, *dt_b_b, *A_log_b, *D_b;
};

static LayerP load_p(void* const* d_in, int o) {
  LayerP p;
  p.ln_g = (const float*)d_in[o + 0];  p.ln_b  = (const float*)d_in[o + 1];
  p.skip = (const float*)d_in[o + 2];
  p.proj_w = (const float*)d_in[o + 3]; p.proj_b = (const float*)d_in[o + 4];
  p.in_w   = (const float*)d_in[o + 5]; p.out_w  = (const float*)d_in[o + 6];
  p.conv_w = (const float*)d_in[o + 7]; p.conv_b = (const float*)d_in[o + 8];
  p.xproj_w = (const float*)d_in[o + 9];
  p.dt_w = (const float*)d_in[o + 10]; p.dt_b = (const float*)d_in[o + 11];
  p.A_log = (const float*)d_in[o + 12]; p.D = (const float*)d_in[o + 13];
  p.conv_w_b = (const float*)d_in[o + 14]; p.conv_b_b = (const float*)d_in[o + 15];
  p.xproj_w_b = (const float*)d_in[o + 16];
  p.dt_w_b = (const float*)d_in[o + 17]; p.dt_b_b = (const float*)d_in[o + 18];
  p.A_log_b = (const float*)d_in[o + 19]; p.D_b = (const float*)d_in[o + 20];
  return p;
}

struct WS {
  bf16_t *wIn[2], *wXp[2][2], *wOut[2], *wProj[2];
  float  *An[2][2];
  float  *h, *xn, *xz, *xc[2], *dbl[2], *y[2], *yo, *l1;
};

static inline void gemm(hipStream_t st, const float* A, const float* A2, const bf16_t* W,
                        float* C, const float* bias, const float* resid, int N, int K,
                        int cm) {
  const int blocks = ((MTOK / 16) * (N / 32)) / 8;   // 8 waves/block, exact
  if (K == 64) {
    k_gemm_bf16<64, false><<<blocks, 256, 0, st>>>(A, nullptr, W, C, bias, resid, N, cm);
  } else if (A2) {
    k_gemm_bf16<128, true><<<blocks, 256, 0, st>>>(A, A2, W, C, bias, resid, N, cm);
  } else {
    k_gemm_bf16<128, false><<<blocks, 256, 0, st>>>(A, nullptr, W, C, bias, resid, N, cm);
  }
}

static void run_layer(hipStream_t st, const float* Xcm, const LayerP& P, WS& w, int li,
                      float* outCM, const float* resid) {
  // LN over tokens
  k_ln_tok<<<MTOK / 8, 256, 0, st>>>(Xcm, nullptr, nullptr, P.ln_g, P.ln_b, w.xn);
  // in_proj: (M,64) x (256,64)^T
  gemm(st, w.xn, nullptr, w.wIn[li], w.xz, nullptr, nullptr, 256, 64, 0);
  // causal depthwise conv + SiLU, both directions
  k_conv_silu<<<(MTOK * DIN) / 256, 256, 0, st>>>(w.xz, P.conv_w, P.conv_b, w.xc[0], +1);
  k_conv_silu<<<(MTOK * DIN) / 256, 256, 0, st>>>(w.xz, P.conv_w_b, P.conv_b_b, w.xc[1], -1);
  // x_proj: (M,128) x (64,128)^T   (rows 36..63 zero-padded)
  gemm(st, w.xc[0], nullptr, w.wXp[li][0], w.dbl[0], nullptr, nullptr, XPAD, 128, 0);
  gemm(st, w.xc[1], nullptr, w.wXp[li][1], w.dbl[1], nullptr, nullptr, XPAD, 128, 0);
  // selective scan (both dirs, one launch)
  ScanArgs sa;
  sa.xc[0] = w.xc[0]; sa.xc[1] = w.xc[1];
  sa.xdbl[0] = w.dbl[0]; sa.xdbl[1] = w.dbl[1];
  sa.dtw[0] = P.dt_w; sa.dtw[1] = P.dt_w_b;
  sa.dtb[0] = P.dt_b; sa.dtb[1] = P.dt_b_b;
  sa.An[0] = w.An[li][0]; sa.An[1] = w.An[li][1];
  sa.Dp[0] = P.D; sa.Dp[1] = P.D_b;
  sa.xz = w.xz;
  sa.yo[0] = w.y[0]; sa.yo[1] = w.y[1];
  dim3 gscan(BATCH, 2);
  k_scan<<<gscan, 512, 0, st>>>(sa);
  // out_proj with fused y_f + y_b: (M,128) x (64,128)^T
  gemm(st, w.y[0], w.y[1], w.wOut[li], w.yo, nullptr, nullptr, 64, 128, 0);
  // LN(yo + skip*xf), same gamma/beta
  k_ln_tok<<<MTOK / 8, 256, 0, st>>>(Xcm, w.yo, P.skip, P.ln_g, P.ln_b, w.xn);
  // proj + bias, transposed store to (B,C,HW), optional residual
  gemm(st, w.xn, nullptr, w.wProj[li], outCM, P.proj_b, resid, 64, 64, 1);
}

extern "C" void kernel_launch(void* const* d_in, const int* in_sizes, int n_in,
                              void* d_out, int out_size, void* d_ws, size_t ws_size,
                              hipStream_t stream) {
  (void)in_sizes; (void)n_in; (void)out_size; (void)ws_size;
  const float* x     = (const float*)d_in[0];
  const float* gn1_g = (const float*)d_in[1];
  const float* gn1_b = (const float*)d_in[2];
  const float* gn2_g = (const float*)d_in[3];
  const float* gn2_b = (const float*)d_in[4];
  LayerP P[2] = {load_p(d_in, 5), load_p(d_in, 26)};

  // ---- carve workspace (~88 MB) ----
  char*  base = (char*)d_ws;
  size_t off  = 0;
  auto alloc = [&](size_t bytes) -> void* {
    void* r = base + off;
    off = (off + bytes + 255) & ~(size_t)255;
    return r;
  };
  WS w;
  for (int li = 0; li < 2; ++li) {
    w.wIn[li]    = (bf16_t*)alloc(256 * 64 * sizeof(bf16_t));
    w.wXp[li][0] = (bf16_t*)alloc(XPAD * 128 * sizeof(bf16_t));
    w.wXp[li][1] = (bf16_t*)alloc(XPAD * 128 * sizeof(bf16_t));
    w.wOut[li]   = (bf16_t*)alloc(64 * 128 * sizeof(bf16_t));
    w.wProj[li]  = (bf16_t*)alloc(64 * 64 * sizeof(bf16_t));
    w.An[li][0]  = (float*)alloc(DIN * DSTATE * sizeof(float));
    w.An[li][1]  = (float*)alloc(DIN * DSTATE * sizeof(float));
  }
  w.h     = (float*)alloc((size_t)MTOK * 64 * sizeof(float));
  w.xn    = (float*)alloc((size_t)MTOK * 64 * sizeof(float));
  w.xz    = (float*)alloc((size_t)MTOK * 256 * sizeof(float));
  w.xc[0] = (float*)alloc((size_t)MTOK * 128 * sizeof(float));
  w.xc[1] = (float*)alloc((size_t)MTOK * 128 * sizeof(float));
  w.dbl[0] = (float*)alloc((size_t)MTOK * XPAD * sizeof(float));
  w.dbl[1] = (float*)alloc((size_t)MTOK * XPAD * sizeof(float));
  w.y[0]  = (float*)alloc((size_t)MTOK * 128 * sizeof(float));
  w.y[1]  = (float*)alloc((size_t)MTOK * 128 * sizeof(float));
  w.yo    = (float*)alloc((size_t)MTOK * 64 * sizeof(float));
  w.l1    = (float*)alloc((size_t)MTOK * 64 * sizeof(float));

  // ---- weight prep (every call; deterministic) ----
  auto pad = [&](const float* s, bf16_t* d, int rows, int cols, int dr) {
    int tot = dr * cols;
    k_pad_bf16<<<(tot + 255) / 256, 256, 0, stream>>>(s, d, rows, cols, dr);
  };
  for (int li = 0; li < 2; ++li) {
    pad(P[li].in_w,      w.wIn[li],    256, 64, 256);
    pad(P[li].xproj_w,   w.wXp[li][0], 36, 128, XPAD);
    pad(P[li].xproj_w_b, w.wXp[li][1], 36, 128, XPAD);
    pad(P[li].out_w,     w.wOut[li],   64, 128, 64);
    pad(P[li].proj_w,    w.wProj[li],  64, 64, 64);
    k_negexp<<<(DIN * DSTATE + 255) / 256, 256, 0, stream>>>(P[li].A_log,   w.An[li][0], DIN * DSTATE);
    k_negexp<<<(DIN * DSTATE + 255) / 256, 256, 0, stream>>>(P[li].A_log_b, w.An[li][1], DIN * DSTATE);
  }

  // ---- block ----
  k_gn_relu<<<BATCH * NGRP, 256, 0, stream>>>(x, gn1_g, gn1_b, w.h);
  run_layer(stream, w.h, P[0], w, 0, w.l1, nullptr);
  k_gn_relu<<<BATCH * NGRP, 256, 0, stream>>>(w.l1, gn2_g, gn2_b, w.h);
  run_layer(stream, w.h, P[1], w, 1, (float*)d_out, x);   // fused +identity residual
}
